// DeepWukong_22857815949595
// MI455X (gfx1250) — matrix operand order repaired and checked by reading, not compiled, not run on hardware
//
#include <hip/hip_runtime.h>
#include <hip/hip_bf16.h>

// ---------------------------------------------------------------------------
// GCN forward for MI455X (gfx1250, wave32, WMMA).
// - Edge gather/scatter phase is L2-resident (xw 83MB + h 80MB < 192MB L2).
// - All GEMMs are memory-bound -> f16 WMMA with f32 accumulate.
// - A and B operands are pre-staged as zero-padded f16 (B transposed) so the
//   WMMA inner loop is pure b128 loads + v_wmma, no bound checks/branches.
// ---------------------------------------------------------------------------

typedef __attribute__((ext_vector_type(16))) _Float16 v16h;
typedef __attribute__((ext_vector_type(8)))  _Float16 v8h;
typedef __attribute__((ext_vector_type(8)))  float    v8f;

// ---------------------------------------------------------------------------
__global__ void fill_f32_kernel(float* __restrict__ p, float v, long long n) {
  long long i = (long long)blockIdx.x * blockDim.x + threadIdx.x;
  if (i < n) p[i] = v;
}

// deg[dst] += 1 for each edge (deg pre-initialized to 1.0 => self loop)
__global__ void deg_kernel(const int* __restrict__ dst, float* __restrict__ deg,
                           int E) {
  int e = blockIdx.x * blockDim.x + threadIdx.x;
  if (e < E) atomicAdd(&deg[dst[e]], 1.0f);
}

__global__ void dinv_kernel(const float* __restrict__ deg,
                            float* __restrict__ dinv, int N) {
  int i = blockIdx.x * blockDim.x + threadIdx.x;
  if (i < N) {
    float d = deg[i];                 // >= 1 always (self loop)
    dinv[i] = (d > 0.0f) ? rsqrtf(d) : 0.0f;
  }
}

// Row-major activation staging: out[M][Kpad] f16, zero-padded in K.
__global__ void pad_a_f16_kernel(const float* __restrict__ in, int lda, int K,
                                 _Float16* __restrict__ out, int Kpad,
                                 long long total) {
  long long i = (long long)blockIdx.x * blockDim.x + threadIdx.x;
  if (i >= total) return;
  int k = (int)(i % Kpad);
  long long m = i / Kpad;
  out[i] = (k < K) ? (_Float16)in[m * lda + k] : (_Float16)0.0f;
}

// Transposed weight staging: out[Npad][Kpad] f16, out[n][k] = B[k][n], padded.
__global__ void pad_bt_f16_kernel(const float* __restrict__ B, int K, int Nn,
                                  _Float16* __restrict__ out, int Kpad,
                                  int Npad) {
  int i = blockIdx.x * blockDim.x + threadIdx.x;
  if (i >= Npad * Kpad) return;
  int k = i % Kpad, n = i / Kpad;
  out[i] = (k < K && n < Nn) ? (_Float16)B[(size_t)k * Nn + n] : (_Float16)0.0f;
}

// ---------------------------------------------------------------------------
// WMMA GEMM body: Out[M x ntilesN*16] = act(Ap[M][Kpad] * Bt^T + bias)
// Ap: f16 row-major, K zero-padded to Kpad (multiple of 32).
// Bt: f16 B-transposed [Npad][Kpad], zero-padded.
// One wave32 per 16x16 tile; Kpad/32 v_wmma_f32_16x16x32_f16 steps.
// Fragment layouts per CDNA5 ISA 7.12.2:
//   A 16x32 f16 : lanes 0-15  -> M=lane,    K = (0..7, 16..23)
//                 lanes 16-31 -> M=lane-16, K = (8..15, 24..31)
//   B 32x16 f16 : lane -> col N=lane&15; lanes 0-15 K=(0..7, 16..23), 16-31 K=(8..15, 24..31)
//   C/D f32     : VGPR v -> row v + 8*(lane>=16), col = lane&15
// All loads are 16B-aligned b128 (Kpad%32==0 guarantees alignment).
// M must be a multiple of 16 (true here: 100000, 512).
// ---------------------------------------------------------------------------
__device__ __forceinline__ void gemm_body(const _Float16* __restrict__ Ap,
                                          const _Float16* __restrict__ Bt,
                                          int Kpad,
                                          const float* __restrict__ bias,
                                          int relu, float* __restrict__ Out,
                                          int ldc, int ntilesN,
                                          int totalTiles) {
  int tile = blockIdx.x * (blockDim.x >> 5) + (threadIdx.x >> 5);
  if (tile >= totalTiles) return;           // uniform per wave: EXEC stays full
  int lane = threadIdx.x & 31;
  int hi   = lane >> 4;                     // 0: lanes 0-15, 1: lanes 16-31
  int l15  = lane & 15;
  int tm = tile / ntilesN, tn = tile % ntilesN;
  int m0 = tm << 4, n0 = tn << 4;

  const _Float16* Ar = Ap + (size_t)(m0 + l15) * Kpad;
  const _Float16* Bc = Bt + (size_t)(n0 + l15) * Kpad;
  int aoff0 = hi ? 8 : 0;                   // A: K block 1
  int aoff1 = hi ? 24 : 16;                 // A: K block 2
  int boff  = hi ? 8 : 0;                   // B: K block 1 (block 2 at +16)

  v8f acc = {};
#pragma unroll
  for (int k0 = 0; k0 < Kpad; k0 += 32) {
    v8h alo = *(const v8h*)(Ar + k0 + aoff0);
    v8h ahi = *(const v8h*)(Ar + k0 + aoff1);
    v8h blo = *(const v8h*)(Bc + k0 + boff);
    v8h bhi = *(const v8h*)(Bc + k0 + boff + 16);
    v16h a = __builtin_shufflevector(alo, ahi, 0, 1, 2, 3, 4, 5, 6, 7, 8, 9,
                                     10, 11, 12, 13, 14, 15);
    v16h bf = __builtin_shufflevector(blo, bhi, 0, 1, 2, 3, 4, 5, 6, 7, 8, 9,
                                      10, 11, 12, 13, 14, 15);
    // 8 args: (neg_a, A, neg_b, B, c_mod, C, reuse_a, reuse_b)
    acc = __builtin_amdgcn_wmma_f32_16x16x32_f16(false, a, false, bf, (short)0,
                                                 acc, false, false);
  }

  int orow = m0 + (hi ? 8 : 0);
  int ocol = n0 + l15;
  float bv = bias ? bias[ocol] : 0.0f;      // MLP GEMMs have N == Npad
  float* op = Out + (size_t)orow * ldc + ocol;
#pragma unroll
  for (int v = 0; v < 8; ++v) {
    float val = acc[v] + bv;
    if (relu) val = fmaxf(val, 0.0f);
    op[(size_t)v * ldc] = val;
  }
}

template <int KPAD>
__global__ void wmma_gemm_tpl(const _Float16* __restrict__ Ap,
                              const _Float16* __restrict__ Bt,
                              const float* __restrict__ bias, int relu,
                              float* __restrict__ Out, int ldc, int ntilesN,
                              int totalTiles) {
  gemm_body(Ap, Bt, KPAD, bias, relu, Out, ldc, ntilesN, totalTiles);
}

__global__ void wmma_gemm_dyn(const _Float16* __restrict__ Ap,
                              const _Float16* __restrict__ Bt, int Kpad,
                              const float* __restrict__ bias, int relu,
                              float* __restrict__ Out, int ldc, int ntilesN,
                              int totalTiles) {
  gemm_body(Ap, Bt, Kpad, bias, relu, Out, ldc, ntilesN, totalTiles);
}

// ---------------------------------------------------------------------------
// h[dst] += xw[src] * dinv[src]*dinv[dst]  (one thread per edge x float4 chunk)
// xw and h are L2-resident on MI455X (83MB + 80MB < 192MB L2).
// ---------------------------------------------------------------------------
__global__ void edge_scatter_kernel(const int* __restrict__ src,
                                    const int* __restrict__ dst,
                                    const float* __restrict__ dinv,
                                    const float* __restrict__ xw,
                                    float* __restrict__ h, int E, int C4,
                                    int xw_ld, int h_ld) {
  long long gid = (long long)blockIdx.x * blockDim.x + threadIdx.x;
  long long tot = (long long)E * C4;
  if (gid >= tot) return;
  int e = (int)(gid / C4);
  int c = (int)(gid % C4);
  int s = src[e], d = dst[e];
  float nm = dinv[s] * dinv[d];
  const float4 v = *(const float4*)(xw + (size_t)s * xw_ld + c * 4);
  float* hp = h + (size_t)d * h_ld + c * 4;
  atomicAdd(hp + 0, v.x * nm);
  atomicAdd(hp + 1, v.y * nm);
  atomicAdd(hp + 2, v.z * nm);
  atomicAdd(hp + 3, v.w * nm);
}

// ---------------------------------------------------------------------------
// pooled[batch[n]] += h[n] + b + xw[n]*dinv[n]^2   (self-loop folded in here)
// batch is sorted -> thread per feature runs over a node chunk and flushes one
// atomic per graph boundary instead of one per node (~200x fewer atomics).
// ---------------------------------------------------------------------------
__global__ void pool_kernel(const float* __restrict__ h,
                            const float* __restrict__ xw,
                            const float* __restrict__ dinv,
                            const float* __restrict__ b,
                            const int* __restrict__ batch,
                            float* __restrict__ pooled, int N, int chunk,
                            int DOUT, int xw_ld) {
  int f = threadIdx.x;
  if (f >= DOUT) return;
  int n0 = blockIdx.x * chunk;
  if (n0 >= N) return;
  int n1 = n0 + chunk;
  if (n1 > N) n1 = N;
  float bf = b[f];
  float sum = 0.0f;
  int g = batch[n0];
  for (int n = n0; n < n1; ++n) {
    int gn = batch[n];
    if (gn != g) {
      atomicAdd(&pooled[(size_t)g * DOUT + f], sum);
      sum = 0.0f;
      g = gn;
    }
    float di = dinv[n];
    sum += h[(size_t)n * DOUT + f] + bf + xw[(size_t)n * xw_ld + f] * di * di;
  }
  atomicAdd(&pooled[(size_t)g * DOUT + f], sum);
}

// ---------------------------------------------------------------------------
// logits = z2 @ Wc + bc; softmax over the 2 classes. N=2 -> scalar is optimal.
// ---------------------------------------------------------------------------
__global__ void classifier_kernel(const float* __restrict__ z2,
                                  const float* __restrict__ Wc,
                                  const float* __restrict__ bc,
                                  float* __restrict__ out, int G, int H,
                                  int z_ld) {
  int g = blockIdx.x * blockDim.x + threadIdx.x;
  if (g >= G) return;
  const float* zg = z2 + (size_t)g * z_ld;
  float l0 = bc[0], l1 = bc[1];
  for (int k = 0; k < H; ++k) {
    float z = zg[k];
    l0 += z * Wc[k * 2 + 0];
    l1 += z * Wc[k * 2 + 1];
  }
  float m = fmaxf(l0, l1);
  float e0 = expf(l0 - m), e1 = expf(l1 - m);
  float s = e0 + e1;
  out[g * 2 + 0] = e0 / s;
  out[g * 2 + 1] = e1 / s;
}

// ---------------------------------------------------------------------------
extern "C" void kernel_launch(void* const* d_in, const int* in_sizes, int n_in,
                              void* d_out, int out_size, void* d_ws,
                              size_t ws_size, hipStream_t stream) {
  const float* x  = (const float*)d_in[0];
  const float* W  = (const float*)d_in[1];
  const float* b  = (const float*)d_in[2];
  const float* W1 = (const float*)d_in[3];
  const float* b1 = (const float*)d_in[4];
  const float* W2 = (const float*)d_in[5];
  const float* b2 = (const float*)d_in[6];
  const float* Wc = (const float*)d_in[7];
  const float* bc = (const float*)d_in[8];
  const int* edge_index = (const int*)d_in[9];
  const int* batch      = (const int*)d_in[10];
  float* out = (float*)d_out;

  const int N    = in_sizes[10];          // 100000
  const int E    = in_sizes[9] / 2;       // 1600000
  const int DIN  = in_sizes[0] / N;       // 100
  const int DOUT = in_sizes[2];           // 200
  const int H    = in_sizes[4];           // 400
  const int G    = out_size / 2;          // 512

  auto up16 = [](int v) { return (v + 15) & ~15; };
  auto up32 = [](int v) { return (v + 31) & ~31; };
  const int DINp    = up32(DIN);    // 128
  const int DOUTp16 = up16(DOUT);   // 208 (xw column pad / stride)
  const int DOUTp32 = up32(DOUT);   // 224 (K pad for MLP GEMM1)
  const int Hp16    = up16(H);      // 400
  const int Hp32    = up32(H);      // 416 (K pad for MLP GEMM2)

  // ---- carve workspace (~192 MB total) --------------------------------
  size_t off = 0;
  auto take = [&](size_t bytes) -> void* {
    void* p = (char*)d_ws + off;
    off += (bytes + 255) & ~(size_t)255;
    return p;
  };
  float*    deg     = (float*)take((size_t)N * 4);
  float*    dinv    = (float*)take((size_t)N * 4);
  _Float16* xp      = (_Float16*)take((size_t)N * DINp * 2);       // x, padded f16
  _Float16* Wt      = (_Float16*)take((size_t)DOUTp16 * DINp * 2); // W^T padded
  _Float16* W1t     = (_Float16*)take((size_t)Hp16 * DOUTp32 * 2); // W1^T padded
  _Float16* W2t     = (_Float16*)take((size_t)Hp16 * Hp32 * 2);    // W2^T padded
  float*    xw      = (float*)take((size_t)N * DOUTp16 * 4);
  float*    h       = (float*)take((size_t)N * DOUT * 4);
  float*    pooled  = (float*)take((size_t)G * DOUT * 4);
  _Float16* pooledp = (_Float16*)take((size_t)G * DOUTp32 * 2);
  float*    z1      = (float*)take((size_t)G * Hp16 * 4);
  _Float16* z1p     = (_Float16*)take((size_t)G * Hp32 * 2);
  float*    z2      = (float*)take((size_t)G * Hp16 * 4);
  (void)ws_size; (void)n_in;

  const int T = 256;
  auto nb = [&](long long n) { return (int)((n + T - 1) / T); };

  auto launch_gemm = [&](const _Float16* Ap, const _Float16* Bt, int Kpad,
                         const float* bias, int relu, float* Out, int ldc,
                         int ntilesN, int tiles) {
    int blocks = (tiles + 7) / 8;           // 8 waves (tiles) per 256-thr block
    switch (Kpad) {
      case 128:
        wmma_gemm_tpl<128><<<blocks, T, 0, stream>>>(Ap, Bt, bias, relu, Out,
                                                     ldc, ntilesN, tiles);
        break;
      case 224:
        wmma_gemm_tpl<224><<<blocks, T, 0, stream>>>(Ap, Bt, bias, relu, Out,
                                                     ldc, ntilesN, tiles);
        break;
      case 416:
        wmma_gemm_tpl<416><<<blocks, T, 0, stream>>>(Ap, Bt, bias, relu, Out,
                                                     ldc, ntilesN, tiles);
        break;
      default:
        wmma_gemm_dyn<<<blocks, T, 0, stream>>>(Ap, Bt, Kpad, bias, relu, Out,
                                                ldc, ntilesN, tiles);
        break;
    }
  };

  // 1) init accumulators (deg=1.0 encodes the self loop)
  fill_f32_kernel<<<nb(N), T, 0, stream>>>(deg, 1.0f, N);
  fill_f32_kernel<<<nb((long long)N * DOUT), T, 0, stream>>>(
      h, 0.0f, (long long)N * DOUT);
  fill_f32_kernel<<<nb((long long)G * DOUT), T, 0, stream>>>(
      pooled, 0.0f, (long long)G * DOUT);

  // 2) degree + dinv
  deg_kernel<<<nb(E), T, 0, stream>>>(edge_index + E, deg, E);
  dinv_kernel<<<nb(N), T, 0, stream>>>(deg, dinv, N);

  // 3) stage padded f16 operands (A row-major, B transposed)
  pad_a_f16_kernel<<<nb((long long)N * DINp), T, 0, stream>>>(
      x, DIN, DIN, xp, DINp, (long long)N * DINp);
  pad_bt_f16_kernel<<<nb(DOUTp16 * DINp), T, 0, stream>>>(W, DIN, DOUT, Wt,
                                                          DINp, DOUTp16);
  pad_bt_f16_kernel<<<nb(Hp16 * DOUTp32), T, 0, stream>>>(W1, DOUT, H, W1t,
                                                          DOUTp32, Hp16);
  pad_bt_f16_kernel<<<nb(Hp16 * Hp32), T, 0, stream>>>(W2, H, H, W2t, Hp32,
                                                       Hp16);

  // 4) xw = x @ W  (WMMA, 6250x13 tiles, 4 k-steps each)
  {
    int mt = N / 16, nt = DOUTp16 / 16, tiles = mt * nt;
    launch_gemm(xp, Wt, DINp, nullptr, 0, xw, DOUTp16, nt, tiles);
  }

  // 5) edge scatter: h[dst] += xw[src] * dinv[src]*dinv[dst]
  {
    int C4 = DOUT / 4;
    long long tot = (long long)E * C4;
    edge_scatter_kernel<<<nb(tot), T, 0, stream>>>(
        edge_index, edge_index + E, dinv, xw, h, E, C4, DOUTp16, DOUT);
  }

  // 6) global add pool (+bias, +self-loop term)
  {
    int chunk = 256;
    pool_kernel<<<(N + chunk - 1) / chunk, T, 0, stream>>>(
        h, xw, dinv, b, batch, pooled, N, chunk, DOUT, DOUTp16);
  }

  // 7) MLP: z1 = relu(pooled@W1+b1); z2 = relu(z1@W2+b2)
  {
    int mt = G / 16, nt = Hp16 / 16, tiles = mt * nt;
    pad_a_f16_kernel<<<nb((long long)G * DOUTp32), T, 0, stream>>>(
        pooled, DOUT, DOUT, pooledp, DOUTp32, (long long)G * DOUTp32);
    launch_gemm(pooledp, W1t, DOUTp32, b1, 1, z1, Hp16, nt, tiles);
    pad_a_f16_kernel<<<nb((long long)G * Hp32), T, 0, stream>>>(
        z1, Hp16, H, z1p, Hp32, (long long)G * Hp32);
    launch_gemm(z1p, W2t, Hp32, b2, 1, z2, Hp16, nt, tiles);
  }

  // 8) classifier + softmax -> d_out [G,2]
  classifier_kernel<<<nb(G), T, 0, stream>>>(z2, Wc, bc, out, G, H, Hp16);
}
